// StructureGraphMessagePassing_12979391168949
// MI455X (gfx1250) — compile-verified, matrix operand order checked by reading, not run on hardware
//
#include <hip/hip_runtime.h>
#include <hip/hip_bf16.h>
#include <cstdint>

typedef __attribute__((ext_vector_type(16))) _Float16 v16h;
typedef __attribute__((ext_vector_type(8)))  float    v8f;

// ---------------------------------------------------------------------------
// WMMA GEMM:  C[M,N] (f32) = A[M,K] (f16, row-major, lda) @ B[N,K]^T (f32 row-major, ldb)
// Requires M%64==0, N%256==0, K%32==0.
// Block = 256 threads (8 waves), tile 64(M) x 256(N), waves 2(m) x 4(n),
// each wave computes 32x64 => 8 accumulators, 8 WMMA per 32-K step.
// ---------------------------------------------------------------------------
#define TM 64
#define TN 256
#define TK 32
#define LSTR 40   // f16 LDS row stride (80B, 16B-aligned rows)

union FragU { uint4 q[2]; v16h h; };

__global__ __launch_bounds__(256)
void gemm_kernel(const _Float16* __restrict__ A, int lda,
                 const float* __restrict__ B, int ldb,
                 float* __restrict__ C, int ldc,
                 int M, int N, int K, const float* __restrict__ bias)
{
    __shared__ __align__(16) _Float16 As[TM][LSTR];
    __shared__ __align__(16) _Float16 Bs[TN][LSTR];

    const int tid  = threadIdx.x;
    const int lane = tid & 31;
    const int wave = tid >> 5;         // 0..7
    const int wm   = wave >> 2;        // 0..1  (32-row strip)
    const int wn   = wave & 3;         // 0..3  (64-col strip)
    const int n0   = blockIdx.x * TN;
    const int m0   = blockIdx.y * TM;

    const int arow = tid >> 2;         // 0..63 (A loader row, 4 threads/row)
    const int aq   = tid & 3;          // 16B quarter within row

    const int frow  = lane & 15;       // fragment row (m or n)
    const int fhalf = lane >> 4;       // fragment k-half (8 elems)

    v8f acc[2][4] = {};

    for (int k0 = 0; k0 < K; k0 += TK) {
        // --- stage A tile 64x32 (f16 -> LDS): 4 threads per row, 8 f16 each ---
        {
            const uint4* src = (const uint4*)(A + (size_t)(m0 + arow) * lda + k0 + aq * 8);
            *(uint4*)&As[arow][aq * 8] = src[0];
        }
        // --- stage B tile 256x32 (f32 -> f16 -> LDS): 1 thread per row, 128B each ---
        {
            const float4* src = (const float4*)(B + (size_t)(n0 + tid) * ldb + k0);
            union { _Float16 h[32]; uint4 q[4]; } t;
            #pragma unroll
            for (int p = 0; p < 8; ++p) {
                float4 f = src[p];
                t.h[p * 4 + 0] = (_Float16)f.x;
                t.h[p * 4 + 1] = (_Float16)f.y;
                t.h[p * 4 + 2] = (_Float16)f.z;
                t.h[p * 4 + 3] = (_Float16)f.w;
            }
            #pragma unroll
            for (int p = 0; p < 4; ++p)
                *(uint4*)&Bs[tid][p * 8] = t.q[p];
        }
        __syncthreads();

        FragU fa[2], fb;
        #pragma unroll
        for (int ms = 0; ms < 2; ++ms) {
            fa[ms].q[0] = *(const uint4*)&As[wm * 32 + ms * 16 + frow][fhalf * 8];
            fa[ms].q[1] = *(const uint4*)&As[wm * 32 + ms * 16 + frow][fhalf * 8 + 16];
        }
        #pragma unroll
        for (int ns = 0; ns < 4; ++ns) {
            fb.q[0] = *(const uint4*)&Bs[wn * 64 + ns * 16 + frow][fhalf * 8];
            fb.q[1] = *(const uint4*)&Bs[wn * 64 + ns * 16 + frow][fhalf * 8 + 16];
            #pragma unroll
            for (int ms = 0; ms < 2; ++ms)
                acc[ms][ns] = __builtin_amdgcn_wmma_f32_16x16x32_f16(
                    false, fa[ms].h, false, fb.h, (short)0, acc[ms][ns], false, false);
        }
        __syncthreads();
    }

    // epilogue: lane L, VGPR v -> m = 8*(L>>4)+v within 16-row subtile, n = L&15
    #pragma unroll
    for (int ms = 0; ms < 2; ++ms) {
        const int mo = m0 + wm * 32 + ms * 16 + fhalf * 8;
        #pragma unroll
        for (int ns = 0; ns < 4; ++ns) {
            const int nc = n0 + wn * 64 + ns * 16 + frow;
            const float bv = bias ? bias[nc] : 0.0f;
            #pragma unroll
            for (int v = 0; v < 8; ++v)
                C[(size_t)(mo + v) * ldc + nc] = acc[ms][ns][v] + bv;
        }
    }
}

// ---------------------------------------------------------------------------
// elementwise / pack / reduce helpers
// ---------------------------------------------------------------------------
__global__ void k_zero(float* p, int total) {
    int i = blockIdx.x * blockDim.x + threadIdx.x;
    if (i < total) p[i] = 0.0f;
}

__global__ void k_scatter_rows(const float* __restrict__ src, const int* __restrict__ idx,
                               float* __restrict__ dst, int D, int total) {
    int i = blockIdx.x * blockDim.x + threadIdx.x;
    if (i >= total) return;
    int e = i / D, c = i % D;
    dst[(size_t)idx[e] * D + c] = src[i];
}

__global__ void k_cvt(const float* __restrict__ x, _Float16* __restrict__ y, int total) {
    int i = blockIdx.x * blockDim.x + threadIdx.x;
    if (i < total) y[i] = (_Float16)x[i];
}

__global__ void k_transpose(const float* __restrict__ in, float* __restrict__ out,
                            int R, int C, int total) {
    int i = blockIdx.x * blockDim.x + threadIdx.x;
    if (i >= total) return;
    int r = i / C, c = i % C;
    out[(size_t)c * R + r] = in[i];
}

// out row-stride 4D: [x, y, x*y, (ymx? y-x : x-y)]
__global__ void k_pack_quad(const float* __restrict__ x, const float* __restrict__ y,
                            _Float16* __restrict__ out, int D, int ymx, int total) {
    int i = blockIdx.x * blockDim.x + threadIdx.x;
    if (i >= total) return;
    int m = i / D, c = i % D;
    float xv = x[i], yv = y[i];
    _Float16* o = out + (size_t)m * 4 * D + c;
    o[0]     = (_Float16)xv;
    o[D]     = (_Float16)yv;
    o[2 * D] = (_Float16)(xv * yv);
    o[3 * D] = (_Float16)(ymx ? (yv - xv) : (xv - yv));
}

__global__ void k_pack2(const float* __restrict__ x, const float* __restrict__ y,
                        _Float16* __restrict__ out, int D, int total) {
    int i = blockIdx.x * blockDim.x + threadIdx.x;
    if (i >= total) return;
    int m = i / D, c = i % D;
    _Float16* o = out + (size_t)m * 2 * D + c;
    o[0] = (_Float16)x[i];
    o[D] = (_Float16)y[i];
}

__global__ void k_pack2_gather(const float* __restrict__ x, const float* __restrict__ y,
                               const int* __restrict__ rows,
                               _Float16* __restrict__ out, int D, int total) {
    int i = blockIdx.x * blockDim.x + threadIdx.x;
    if (i >= total) return;
    int m = i / D, c = i % D;
    _Float16* o = out + (size_t)m * 2 * D + c;
    o[0] = (_Float16)x[i];
    o[D] = (_Float16)y[(size_t)rows[m] * D + c];
}

__global__ void k_pack3(const float* __restrict__ x, const float* __restrict__ y,
                        const float* __restrict__ z,
                        _Float16* __restrict__ out, int D, int total) {
    int i = blockIdx.x * blockDim.x + threadIdx.x;
    if (i >= total) return;
    int m = i / D, c = i % D;
    _Float16* o = out + (size_t)m * 3 * D + c;
    o[0]     = (_Float16)x[i];
    o[D]     = (_Float16)y[i];
    o[2 * D] = (_Float16)z[i];
}

// rel_j[i,j,:] += vjs[i,:] + vjo[j,:]   (in place over rjr)
__global__ void k_relj_combine(float* __restrict__ relj, const float* __restrict__ vjs,
                               const float* __restrict__ vjo, int n, int D, int total) {
    int i = blockIdx.x * blockDim.x + threadIdx.x;
    if (i >= total) return;
    int c  = i % D;
    int ij = i / D;
    int ii = ij / n, jj = ij % n;
    relj[i] = relj[i] + vjs[(size_t)ii * D + c] + vjo[(size_t)jj * D + c];
}

// s_sbj[i,j] = <qs2[i], rel_j[i,j]> * inv_scale ; s_obj[i,j] = <qo2[j], rel_j[i,j]> * inv_scale
__global__ void k_scores(const float* __restrict__ qs2, const float* __restrict__ qo2,
                         const float* __restrict__ relj, int n, int D, float inv_scale,
                         float* __restrict__ s_sbj, float* __restrict__ s_obj) {
    int ij = blockIdx.x;
    int i = ij / n, j = ij % n;
    const float* r = relj + (size_t)ij * D;
    const float* a = qs2 + (size_t)i * D;
    const float* b = qo2 + (size_t)j * D;
    float p1 = 0.f, p2 = 0.f;
    for (int c = threadIdx.x; c < D; c += blockDim.x) {
        float rv = r[c];
        p1 += a[c] * rv;
        p2 += b[c] * rv;
    }
    __shared__ float s1[256], s2[256];
    int t = threadIdx.x;
    s1[t] = p1; s2[t] = p2;
    __syncthreads();
    for (int st = 128; st > 0; st >>= 1) {
        if (t < st) { s1[t] += s1[t + st]; s2[t] += s2[t + st]; }
        __syncthreads();
    }
    if (t == 0) {
        s_sbj[ij] = s1[0] * inv_scale;
        s_obj[ij] = s2[0] * inv_scale;
    }
}

// blocks [0,n): row softmax of s_sbj (axis=1). blocks [n,2n): col softmax of s_obj (axis=0).
__global__ void k_softmax(const float* __restrict__ s_sbj, const float* __restrict__ s_obj,
                          const int* __restrict__ conn, float* __restrict__ w_sbj,
                          float* __restrict__ w_obj, int n) {
    int b = blockIdx.x, t = threadIdx.x;
    bool isRow = (b < n);
    int line = isRow ? b : b - n;
    int idx  = isRow ? line * n + t : t * n + line;
    bool con = conn[idx] >= 0;
    float v = con ? (isRow ? s_sbj[idx] : s_obj[idx]) : -__builtin_inff();
    __shared__ float buf[64];
    buf[t] = v; __syncthreads();
    for (int st = 32; st > 0; st >>= 1) { if (t < st) buf[t] = fmaxf(buf[t], buf[t + st]); __syncthreads(); }
    float mx = buf[0]; __syncthreads();
    float e = con ? expf(v - mx) : 0.0f;
    buf[t] = e; __syncthreads();
    for (int st = 32; st > 0; st >>= 1) { if (t < st) buf[t] += buf[t + st]; __syncthreads(); }
    float w = e / buf[0];
    if (isRow) w_sbj[idx] = w; else w_obj[idx] = w;
}

// z=0: ctx_s[i,d] = sum_j w_sbj[i,j]*relj[i,j,d] ; z=1: ctx_o[j,d] = sum_i w_obj[i,j]*relj[i,j,d]
__global__ void k_ctx(const float* __restrict__ relj, const float* __restrict__ ws,
                      const float* __restrict__ wo, float* __restrict__ ctx_s,
                      float* __restrict__ ctx_o, int n, int D) {
    int c = blockIdx.x * blockDim.x + threadIdx.x;
    int line = blockIdx.y;
    if (blockIdx.z == 0) {
        float acc = 0.f;
        for (int j = 0; j < n; ++j)
            acc += ws[line * n + j] * relj[(size_t)(line * n + j) * D + c];
        ctx_s[(size_t)line * D + c] = acc;
    } else {
        float acc = 0.f;
        for (int i = 0; i < n; ++i)
            acc += wo[i * n + line] * relj[(size_t)(i * n + line) * D + c];
        ctx_o[(size_t)line * D + c] = acc;
    }
}

__global__ void k_vout_fix(const int* __restrict__ conn, const float* __restrict__ visual,
                           float* __restrict__ vout, int n, int D) {
    int i = blockIdx.x;
    __shared__ int sdeg;
    if (threadIdx.x == 0) {
        int dg = 0;
        for (int j = 0; j < n; ++j) dg += (conn[i * n + j] >= 0) + (conn[j * n + i] >= 0);
        sdeg = dg;
    }
    __syncthreads();
    if (sdeg == 0)
        for (int c = threadIdx.x; c < D; c += blockDim.x)
            vout[(size_t)i * D + c] = visual[(size_t)i * D + c];
}

// ---------------------------------------------------------------------------
extern "C" void kernel_launch(void* const* d_in, const int* in_sizes, int n_in,
                              void* d_out, int out_size, void* d_ws, size_t ws_size,
                              hipStream_t stream) {
    (void)in_sizes; (void)n_in; (void)out_size; (void)ws_size;
    constexpr int Nn = 64, Dd = 1024, NN = Nn * Nn, Ee = 1024;

    const float* visual = (const float*)d_in[0];
    const float* relv   = (const float*)d_in[1];
    const int*   conn   = (const int*)d_in[2];
    const int*   eidx   = (const int*)d_in[4];
    const float* W_sub  = (const float*)d_in[5];
    const float* W_obj  = (const float*)d_in[6];
    const float* W_r2s  = (const float*)d_in[7];
    const float* W_r2o  = (const float*)d_in[8];
    const float* W_joint= (const float*)d_in[9];
    const float* W_ctxw = (const float*)d_in[10];
    const float* W_ru   = (const float*)d_in[11];
    const float* W_rjw  = (const float*)d_in[12];
    const float* W_rcw  = (const float*)d_in[13];
    const float* W_node = (const float*)d_in[14];
    const float* b_node = (const float*)d_in[15];
    const float* W_fact = (const float*)d_in[16];

    char* ws = (char*)d_ws;
    size_t off = 0;
    auto alloc = [&](size_t bytes) -> char* {
        char* p = ws + off;
        off += (bytes + 255) & ~(size_t)255;
        return p;
    };
    float*    rel_temp = (float*)alloc((size_t)NN * Dd * 4);
    float*    rel_ctxb = (float*)alloc((size_t)NN * Dd * 4);
    float*    rel_j    = (float*)alloc((size_t)NN * Dd * 4);
    float*    rj       = (float*)alloc((size_t)NN * Dd * 4);
    _Float16* rj16     = (_Float16*)alloc((size_t)NN * Dd * 2);
    _Float16* relcat   = (_Float16*)alloc((size_t)NN * 4 * Dd * 2);
    float*    wt_r2s   = (float*)alloc((size_t)Dd * Dd * 4);
    float*    wt_r2o   = (float*)alloc((size_t)Dd * Dd * 4);
    _Float16* packE    = (_Float16*)alloc((size_t)Ee * 2 * Dd * 2);
    float*    vj       = (float*)alloc((size_t)Nn * Dd * 4);
    float*    vjs      = (float*)alloc((size_t)Nn * Dd * 4);
    float*    vjo      = (float*)alloc((size_t)Nn * Dd * 4);
    float*    qtmp     = (float*)alloc((size_t)Nn * Dd * 4);
    float*    qs2      = (float*)alloc((size_t)Nn * Dd * 4);
    float*    qo2      = (float*)alloc((size_t)Nn * Dd * 4);
    float*    ctx_s    = (float*)alloc((size_t)Nn * Dd * 4);
    float*    ctx_o    = (float*)alloc((size_t)Nn * Dd * 4);
    float*    vctxb    = (float*)alloc((size_t)Nn * Dd * 4);
    _Float16* vj16     = (_Float16*)alloc((size_t)Nn * Dd * 2);
    _Float16* q16      = (_Float16*)alloc((size_t)Nn * Dd * 2);
    _Float16* vcat     = (_Float16*)alloc((size_t)Nn * 4 * Dd * 2);
    float*    s_sbj    = (float*)alloc((size_t)NN * 4);
    float*    s_obj    = (float*)alloc((size_t)NN * 4);
    float*    w_sbj    = (float*)alloc((size_t)NN * 4);
    float*    w_obj    = (float*)alloc((size_t)NN * 4);

    float* rel_out = (float*)d_out;                    // [E, D]
    float* v_out   = (float*)d_out + (size_t)Ee * Dd;  // [N, D]

    auto grid1 = [](int total) { return dim3((unsigned)((total + 255) / 256)); };
    auto gemm = [&](const _Float16* A, int lda, const float* B, int ldb, float* C,
                    int M, int Ncols, int K, const float* bias) {
        gemm_kernel<<<dim3(Ncols / TN, M / TM), 256, 0, stream>>>(
            A, lda, B, ldb, C, Ncols, M, Ncols, K, bias);
    };

    // --- static precompute ---
    k_zero<<<grid1(NN * Dd), 256, 0, stream>>>(rel_temp, NN * Dd);
    k_scatter_rows<<<grid1(Ee * Dd), 256, 0, stream>>>(relv, eidx, rel_temp, Dd, Ee * Dd);
    k_transpose<<<grid1(Dd * Dd), 256, 0, stream>>>(W_r2s, wt_r2s, Dd, Dd, Dd * Dd);
    k_transpose<<<grid1(Dd * Dd), 256, 0, stream>>>(W_r2o, wt_r2o, Dd, Dd, Dd * Dd);

    const float* vctx_cur = visual;
    const float* rctx_cur = rel_temp;
    const float inv_scale = 1.0f / 32.0f;  // 1/sqrt(1024)

    for (int t = 0; t < 2; ++t) {
        // vj = [v, vctx, v*vctx, vctx-v] @ W_joint^T
        k_pack_quad<<<grid1(Nn * Dd), 256, 0, stream>>>(visual, vctx_cur, vcat, Dd, 1, Nn * Dd);
        gemm(vcat, 4 * Dd, W_joint, 4 * Dd, vj, Nn, Dd, 4 * Dd, nullptr);
        k_cvt<<<grid1(Nn * Dd), 256, 0, stream>>>(vj, vj16, Nn * Dd);

        // rj = [rt, rc, rt*rc, rt-rc] @ W_rel_joint^T
        k_pack_quad<<<grid1(NN * Dd), 256, 0, stream>>>(rel_temp, rctx_cur, relcat, Dd, 0, NN * Dd);
        gemm(relcat, 4 * Dd, W_rjw, 4 * Dd, rj, NN, Dd, 4 * Dd, nullptr);

        // vjs = vj@Ws^T, vjo = vj@Wo^T  (Ws/Wo/Wr = column slices of W_rel_update)
        gemm(vj16, Dd, W_ru,          3 * Dd, vjs, Nn, Dd, Dd, nullptr);
        gemm(vj16, Dd, W_ru + Dd,     3 * Dd, vjo, Nn, Dd, Dd, nullptr);

        // rel_j = rj@Wr^T + vjs[i] + vjo[j]
        k_cvt<<<grid1(NN * Dd), 256, 0, stream>>>(rj, rj16, NN * Dd);
        gemm(rj16, Dd, W_ru + 2 * Dd, 3 * Dd, rel_j, NN, Dd, Dd, nullptr);
        k_relj_combine<<<grid1(NN * Dd), 256, 0, stream>>>(rel_j, vjs, vjo, Nn, Dd, NN * Dd);

        // qs2 = (vj@W_sub^T)@W_r2s ; qo2 = (vj@W_obj^T)@W_r2o  (associativity hoist)
        gemm(vj16, Dd, W_sub, Dd, qtmp, Nn, Dd, Dd, nullptr);
        k_cvt<<<grid1(Nn * Dd), 256, 0, stream>>>(qtmp, q16, Nn * Dd);
        gemm(q16, Dd, wt_r2s, Dd, qs2, Nn, Dd, Dd, nullptr);
        gemm(vj16, Dd, W_obj, Dd, qtmp, Nn, Dd, Dd, nullptr);
        k_cvt<<<grid1(Nn * Dd), 256, 0, stream>>>(qtmp, q16, Nn * Dd);
        gemm(q16, Dd, wt_r2o, Dd, qo2, Nn, Dd, Dd, nullptr);

        // attention scores + masked softmaxes + context aggregation
        k_scores<<<NN, 256, 0, stream>>>(qs2, qo2, rel_j, Nn, Dd, inv_scale, s_sbj, s_obj);
        k_softmax<<<2 * Nn, 64, 0, stream>>>(s_sbj, s_obj, conn, w_sbj, w_obj, Nn);
        k_ctx<<<dim3(Dd / 256, Nn, 2), 256, 0, stream>>>(rel_j, w_sbj, w_obj, ctx_s, ctx_o, Nn, Dd);

        // vctx = [vctx, ctx_s, ctx_o] @ W_ctx^T
        k_pack3<<<grid1(Nn * Dd), 256, 0, stream>>>(vctx_cur, ctx_s, ctx_o, vcat, Dd, Nn * Dd);
        gemm(vcat, 3 * Dd, W_ctxw, 3 * Dd, vctxb, Nn, Dd, 3 * Dd, nullptr);
        vctx_cur = vctxb;

        // rel_ctx = [rel_ctx, rel_j] @ W_rel_ctx^T
        k_pack2<<<grid1(NN * Dd), 256, 0, stream>>>(rctx_cur, rel_j, relcat, Dd, NN * Dd);
        gemm(relcat, 2 * Dd, W_rcw, 2 * Dd, rel_ctxb, NN, Dd, 2 * Dd, nullptr);
        rctx_cur = rel_ctxb;
    }

    // rel_out = [rel_visual, rel_ctx[edge_idx]] @ W_factor^T
    k_pack2_gather<<<grid1(Ee * Dd), 256, 0, stream>>>(relv, rctx_cur, eidx, packE, Dd, Ee * Dd);
    gemm(packE, 2 * Dd, W_fact, 2 * Dd, rel_out, Ee, Dd, 2 * Dd, nullptr);

    // v_out = [visual, vctx] @ W_node^T + b_node, then deg==0 passthrough
    k_pack2<<<grid1(Nn * Dd), 256, 0, stream>>>(visual, vctx_cur, vcat, Dd, Nn * Dd);
    gemm(vcat, 2 * Dd, W_node, 2 * Dd, v_out, Nn, Dd, 2 * Dd, b_node);
    k_vout_fix<<<Nn, 256, 0, stream>>>(conn, visual, v_out, Nn, Dd);
}
